// DLRMTower_23218593202348
// MI455X (gfx1250) — matrix operand order repaired and checked by reading, not compiled, not run on hardware
//
#include <hip/hip_runtime.h>
#include <hip/hip_bf16.h>
#include <stdint.h>

typedef __attribute__((ext_vector_type(16))) __bf16 v16bf;
typedef __attribute__((ext_vector_type(8)))  float  v8f;
typedef __attribute__((ext_vector_type(4)))  int    v4i;

#define BROWS 16384
#define NCAT  26
#define VOCAB 50000
#define DIMD  128

union Frag { v16bf v; uint4 u[2]; };

static __device__ __forceinline__ __bf16 f2b(float f) { return (__bf16)f; }

#if defined(__HIP_DEVICE_COMPILE__) && __has_builtin(__builtin_amdgcn_global_load_async_to_lds_b128)
#define USE_ASYNC_LDS 1
typedef __attribute__((address_space(1))) v4i* gv4i_p;   // global int4*
typedef __attribute__((address_space(3))) v4i* lv4i_p;   // LDS int4*
static __device__ __forceinline__ gv4i_p as_g(const void* p) {
  return (gv4i_p)(uintptr_t)p;
}
static __device__ __forceinline__ lv4i_p as_l(void* p) {
  return (lv4i_p)(uint32_t)(uintptr_t)p;   // low 32 bits of generic addr = LDS offset
}
static __device__ __forceinline__ void wait_async0() {
#if __has_builtin(__builtin_amdgcn_s_wait_asynccnt)
  __builtin_amdgcn_s_wait_asynccnt(0);
#else
  asm volatile("s_wait_asynccnt 0" ::: "memory");
#endif
}
#else
#define USE_ASYNC_LDS 0
#endif

// ---------------------------------------------------------------------------
// Kernel 1: pad fp32 dense (B x 13) into bf16 (B x 32), zero-padded K.
// ---------------------------------------------------------------------------
__global__ __launch_bounds__(256) void pad_dense_kernel(
    const float* __restrict__ dense, __bf16* __restrict__ A0) {
  int i = blockIdx.x * 256 + threadIdx.x;   // over B*32
  int b = i >> 5, k = i & 31;
  float v = (k < 13) ? dense[b * 13 + k] : 0.0f;
  A0[i] = f2b(v);
}

// ---------------------------------------------------------------------------
// Kernel 2: generic tiled bf16 WMMA GEMM:  out = act(A[MxK] * W[KxN] + bias)
//   Double-buffered LDS; A tile staged via GLOBAL_LOAD_ASYNC_TO_LDS_B128
//   (ASYNCcnt) when available, W tile converted fp32->bf16 through VGPRs.
//   256 threads = 8 waves; block tile 128x64; wave: 16 rows x 4 col-tiles.
// ---------------------------------------------------------------------------
template <bool RELU, bool F32OUT>
__global__ __launch_bounds__(256) void gemm_bias_act_kernel(
    const __bf16* __restrict__ A, int lda,
    const float* __restrict__ W, int K, int N,
    const float* __restrict__ bias,
    __bf16* __restrict__ outb, float* __restrict__ outf) {
  __shared__ __bf16 As[2][128][40];   // 80B pitch -> 20-bank stride, 16B aligned
  __shared__ __bf16 Ws[2][64][40];    // W^T tile: [n][k]

  const int m0 = blockIdx.x * 128;
  const int n0 = blockIdx.y * 64;
  const int t = threadIdx.x;
  const int lane = t & 31;
  const int wid = t >> 5;
  const int KT = (K + 31) / 32;

  // staging coordinates (per thread)
  const int ar = t >> 1;            // A row 0..127
  const int ah = (t & 1) * 16;      // A col half: 0 or 16
  const int wn = t & 63;            // W^T row (output col) 0..63
  const int wks = (t >> 6) * 8;     // W^T k segment: 0,8,16,24

  // fragment coordinates
  const int arow = wid * 16 + (lane & 15);
  const int kbA = (lane >> 4) * 8;     // A fragment K sub-base (ISA 7.12.2)
  const int kbB = (lane >> 4) * 8;     // B fragment K sub-base (ISA 7.12.2)
  const int bcol = lane & 15;

  v8f acc[4] = {};

  auto stage = [&](int kt, int buf) {
    const int k0 = kt * 32;
    const __bf16* srcA = A + (size_t)(m0 + ar) * lda + (k0 + ah);
#if USE_ASYNC_LDS
    __builtin_amdgcn_global_load_async_to_lds_b128(as_g(srcA),
                                                   as_l(&As[buf][ar][ah]), 0, 0);
    __builtin_amdgcn_global_load_async_to_lds_b128(as_g(srcA + 8),
                                                   as_l(&As[buf][ar][ah + 8]), 0, 0);
#else
    uint4 u0 = ((const uint4*)srcA)[0];
    uint4 u1 = ((const uint4*)srcA)[1];
    *(uint4*)&As[buf][ar][ah] = u0;
    *(uint4*)&As[buf][ar][ah + 8] = u1;
#endif
    __attribute__((aligned(16))) __bf16 tmp[8];
    #pragma unroll
    for (int j = 0; j < 8; ++j) {
      int k = k0 + wks + j;
      float v = (k < K) ? W[(size_t)k * N + (n0 + wn)] : 0.0f;
      tmp[j] = f2b(v);
    }
    *(uint4*)&Ws[buf][wn][wks] = *(const uint4*)tmp;
  };

  auto fence_stage = [&]() {
#if USE_ASYNC_LDS
    wait_async0();
#endif
    __syncthreads();
  };

  stage(0, 0);
  fence_stage();

  for (int kt = 0; kt < KT; ++kt) {
    const int cur = kt & 1;
    if (kt + 1 < KT) stage(kt + 1, cur ^ 1);

    Frag af;
    af.u[0] = *(const uint4*)&As[cur][arow][kbA];
    af.u[1] = *(const uint4*)&As[cur][arow][kbA + 16];
    #pragma unroll
    for (int j = 0; j < 4; ++j) {
      Frag bfm;
      const __bf16* bp_ = &Ws[cur][j * 16 + bcol][kbB];
      bfm.u[0] = ((const uint4*)bp_)[0];
      bfm.u[1] = *(const uint4*)(bp_ + 16);
      acc[j] = __builtin_amdgcn_wmma_f32_16x16x32_bf16(
          false, af.v, false, bfm.v, (short)0, acc[j], false, false);
    }
    if (kt + 1 < KT) fence_stage();
  }

  // --- epilogue: bias + activation + store (C/D layout per ISA 7.12.2) ---
  const int rbase = m0 + wid * 16 + ((lane >> 4) << 3);
  #pragma unroll
  for (int j = 0; j < 4; ++j) {
    int col = n0 + j * 16 + (lane & 15);
    float bb = bias[col];
    #pragma unroll
    for (int i = 0; i < 8; ++i) {
      float v = acc[j][i] + bb;
      if (RELU) v = fmaxf(v, 0.0f);
      size_t off = (size_t)(rbase + i) * N + col;
      if (F32OUT) outf[off] = v;
      else        outb[off] = f2b(v);
    }
  }
}

// ---------------------------------------------------------------------------
// Kernel 3: per-batch-row embedding gather + pairwise interaction.
//   One wave32 per batch row; T (32x128 bf16, rows 27..31 zero) in LDS.
//   Z = T * T^T ; only tiles (0,0),(0,1),(1,1) carry m<n pairs -> 12 WMMAs.
//   All 26 row gathers prefetched up front (gather is the HBM roofline).
// ---------------------------------------------------------------------------
__global__ __launch_bounds__(128) void embed_interact_kernel(
    const __bf16* __restrict__ h2,
    const int* __restrict__ idx,          // (26, B)
    const float* __restrict__ tables,     // (26, 50000, 128)
    __bf16* __restrict__ combined) {      // (B, 512)
  __shared__ __bf16 T[4][32][136];        // 272B pitch: 4-bank row stride

  const int t = threadIdx.x;
  const int lane = t & 31;
  const int w = t >> 5;
  const int b = blockIdx.x * 4 + w;
  const int c4 = lane * 4;                // 4 elements / lane per row

  // indices first, then prefetch every gathered row before any dependent work
  int inds[NCAT];
  #pragma unroll
  for (int c = 0; c < NCAT; ++c) inds[c] = idx[c * BROWS + b];
  #pragma unroll
  for (int c = 0; c < NCAT; ++c)
    __builtin_prefetch(tables + ((size_t)c * VOCAB + (size_t)inds[c]) * DIMD + c4, 0, 3);

  // row 0: h2 (already bf16)
  {
    uint2 u = *(const uint2*)(h2 + (size_t)b * DIMD + c4);
    *(uint2*)&T[w][0][c4] = u;
  }
  // rows 1..26: gathered embeddings, fp32 -> bf16
  #pragma unroll 2
  for (int c = 0; c < NCAT; ++c) {
    size_t row = (size_t)c * VOCAB + (size_t)inds[c];
    float4 v = *(const float4*)(tables + row * DIMD + c4);
    __attribute__((aligned(8))) __bf16 o[4] = {f2b(v.x), f2b(v.y), f2b(v.z), f2b(v.w)};
    *(uint2*)&T[w][c + 1][c4] = *(const uint2*)o;
  }
  // rows 27..31: zero pad
  for (int r = 27; r < 32; ++r) {
    uint2 z = make_uint2(0u, 0u);
    *(uint2*)&T[w][r][c4] = z;
  }
  __syncthreads();

  v8f acc00 = {}, acc01 = {}, acc11 = {};
  const int rlo = lane & 15;
  const int kbA = (lane >> 4) * 8;
  const int kbB = (lane >> 4) * 8;

  #pragma unroll
  for (int kt = 0; kt < 4; ++kt) {
    const int k0 = kt * 32;
    Frag a0, a1, b0, b1;
    {
      const __bf16* p = &T[w][rlo][k0 + kbA];
      a0.u[0] = ((const uint4*)p)[0];
      a0.u[1] = *(const uint4*)(p + 16);
    }
    {
      const __bf16* p = &T[w][16 + rlo][k0 + kbA];
      a1.u[0] = ((const uint4*)p)[0];
      a1.u[1] = *(const uint4*)(p + 16);
    }
    {
      const __bf16* p = &T[w][rlo][k0 + kbB];        // B[k][n] = T[n][k]
      b0.u[0] = ((const uint4*)p)[0];
      b0.u[1] = *(const uint4*)(p + 16);
    }
    {
      const __bf16* p = &T[w][16 + rlo][k0 + kbB];
      b1.u[0] = ((const uint4*)p)[0];
      b1.u[1] = *(const uint4*)(p + 16);
    }
    acc00 = __builtin_amdgcn_wmma_f32_16x16x32_bf16(false, a0.v, false, b0.v,
                                                    (short)0, acc00, false, false);
    acc01 = __builtin_amdgcn_wmma_f32_16x16x32_bf16(false, a0.v, false, b1.v,
                                                    (short)0, acc01, false, false);
    acc11 = __builtin_amdgcn_wmma_f32_16x16x32_bf16(false, a1.v, false, b1.v,
                                                    (short)0, acc11, false, false);
  }

  // combined[b][0:128] = h2
  {
    uint2 u = *(const uint2*)&T[w][0][c4];
    *(uint2*)(combined + (size_t)b * 512 + c4) = u;
  }
  // combined[b][128 + triu_idx(m,n)] = Z[m][n], m < n < 27
  __bf16* zout = combined + (size_t)b * 512 + 128;
  const int nlo = lane & 15;
  const int mhi = (lane >> 4) << 3;
  auto emit = [&](const v8f& acc, int mt, int nt) {
    int n = nt * 16 + nlo;
    int mb = mt * 16 + mhi;
    #pragma unroll
    for (int i = 0; i < 8; ++i) {
      int m = mb + i;
      if (m < 27 && n < 27 && m < n) {
        int fidx = 26 * m - (m * (m - 1)) / 2 + (n - m - 1);
        zout[fidx] = f2b(acc[i]);
      }
    }
  };
  emit(acc00, 0, 0);
  emit(acc01, 0, 1);
  emit(acc11, 1, 1);
}

// ---------------------------------------------------------------------------
extern "C" void kernel_launch(void* const* d_in, const int* in_sizes, int n_in,
                              void* d_out, int out_size, void* d_ws, size_t ws_size,
                              hipStream_t stream) {
  (void)in_sizes; (void)n_in; (void)out_size; (void)ws_size;
  const float* dense  = (const float*)d_in[0];
  const int*   idx    = (const int*)d_in[1];
  const float* W0     = (const float*)d_in[2];
  const float* b0     = (const float*)d_in[3];
  const float* W1     = (const float*)d_in[4];
  const float* b1     = (const float*)d_in[5];
  const float* W2     = (const float*)d_in[6];
  const float* b2     = (const float*)d_in[7];
  const float* tables = (const float*)d_in[8];
  const float* Wp     = (const float*)d_in[9];
  const float* bp     = (const float*)d_in[10];
  float* out = (float*)d_out;

  char* ws = (char*)d_ws;
  __bf16* A0   = (__bf16*)(ws);                                   // B*32
  __bf16* h0   = (__bf16*)(ws + (size_t)1  * 1024 * 1024);        // B*512
  __bf16* h1   = (__bf16*)(ws + (size_t)17 * 1024 * 1024);        // B*256
  __bf16* h2   = (__bf16*)(ws + (size_t)25 * 1024 * 1024);        // B*128
  __bf16* comb = (__bf16*)(ws + (size_t)29 * 1024 * 1024);        // B*512

  // 1) pad dense to bf16 (B x 32)
  pad_dense_kernel<<<(BROWS * 32) / 256, 256, 0, stream>>>(dense, A0);
  // 2) MLP layer 0: (B x 13) * (13 x 512), relu
  gemm_bias_act_kernel<true, false><<<dim3(BROWS / 128, 512 / 64), 256, 0, stream>>>(
      A0, 32, W0, 13, 512, b0, h0, nullptr);
  // 3) MLP layer 1: (B x 512) * (512 x 256), relu
  gemm_bias_act_kernel<true, false><<<dim3(BROWS / 128, 256 / 64), 256, 0, stream>>>(
      h0, 512, W1, 512, 256, b1, h1, nullptr);
  // 4) MLP layer 2: (B x 256) * (256 x 128), linear
  gemm_bias_act_kernel<false, false><<<dim3(BROWS / 128, 128 / 64), 256, 0, stream>>>(
      h1, 256, W2, 256, 128, b2, h2, nullptr);
  // 5) embedding gather + self-interaction -> combined (B x 512, K=479 valid)
  embed_interact_kernel<<<BROWS / 4, 128, 0, stream>>>(h2, idx, tables, comb);
  // 6) projection: (B x 479) * (479 x 512) -> fp32 out
  gemm_bias_act_kernel<false, true><<<dim3(BROWS / 128, 512 / 64), 256, 0, stream>>>(
      comb, 512, Wp, 479, 512, bp, nullptr, out);
}
